// MultiHeadSelfAttention_5781025980564
// MI455X (gfx1250) — matrix operand order repaired and checked by reading, not compiled, not run on hardware
//
#include <hip/hip_runtime.h>
#include <hip/hip_bf16.h>

// MI455X / gfx1250 fused multi-head self-attention.
// bf16 WMMA (16x16x32) with f32 accumulation; flash-attention style streaming
// so the [B,H,S,S] score tensor is never materialized in HBM.
// GEMM tiles staged via gfx1250 async-to-LDS (ASYNCcnt); attention K tiles
// staged via the Tensor Data Mover (TENSORcnt).

typedef __bf16 v16bf __attribute__((ext_vector_type(16)));
typedef __bf16 v8bf  __attribute__((ext_vector_type(8)));
typedef float  v8f   __attribute__((ext_vector_type(8)));
typedef unsigned v4u __attribute__((ext_vector_type(4)));
typedef unsigned v8u __attribute__((ext_vector_type(8)));

static __device__ __forceinline__ v8f vzero8() {
    v8f z;
#pragma unroll
    for (int i = 0; i < 8; ++i) z[i] = 0.0f;
    return z;
}

static __device__ __forceinline__ v16bf cat8(v8bf lo, v8bf hi) {
    return __builtin_shufflevector(lo, hi, 0, 1, 2, 3, 4, 5, 6, 7,
                                           8, 9, 10, 11, 12, 13, 14, 15);
}

// gfx1250 async global->LDS copy, 16 bytes per lane. LDS operand is the
// 32-bit LDS byte address (low 32 bits of the flat shared address).
static __device__ __forceinline__ void async_load_b128(const void* gptr, void* lptr) {
    unsigned lds = (unsigned)(size_t)lptr;
    asm volatile("global_load_async_to_lds_b128 %0, %1, off"
                 :: "v"(lds), "v"(gptr) : "memory");
}
static __device__ __forceinline__ void wait_async0() {
    asm volatile("s_wait_asynccnt 0x0" ::: "memory");
}

// Tensor Data Mover: 2D tile load global->LDS (CDNA5 ISA).
// tile_d0 elements contiguous per row, tile_d1 rows, row stride in elements,
// 2-byte elements, tile packed densely into LDS at lds_addr.
static __device__ __forceinline__ void tdm_load_2d_bf16(
    unsigned lds_addr, const void* gaddr,
    unsigned tile_d0, unsigned tile_d1, unsigned stride_elems) {
    unsigned long long ga = (unsigned long long)(size_t)gaddr;
    v4u g0;
    g0[0] = 1u;                                   // count=1, user descriptor
    g0[1] = lds_addr;                             // LDS byte address
    g0[2] = (unsigned)ga;                         // global_addr[31:0]
    g0[3] = (unsigned)(ga >> 32) | (2u << 30);    // global_addr[56:32] | type=2
    v8u g1;
    g1[0] = 1u << 16;                             // data_size=1 (2 bytes)
    g1[1] = tile_d0 << 16;                        // tensor_dim0[15:0] (=tile)
    g1[2] = tile_d1 << 16;                        // tensor_dim0 hi | tensor_dim1 lo
    g1[3] = tile_d0 << 16;                        // tensor_dim1 hi | tile_dim0
    g1[4] = tile_d1;                              // tile_dim1 | tile_dim2=0
    g1[5] = stride_elems;                         // tensor_dim0_stride[31:0]
    g1[6] = 0u;                                   // stride hi | dim1_stride lo
    g1[7] = 0u;                                   // dim1_stride hi
    v4u gz;
    g0 = g0; gz[0] = 0u; gz[1] = 0u; gz[2] = 0u; gz[3] = 0u;
    asm volatile("tensor_load_to_lds %0, %1, %2, %3"
                 :: "s"(g0), "s"(g1), "s"(gz), "s"(gz)
                 : "memory");
}

// ---------------------------------------------------------------- conversions
__global__ __launch_bounds__(256) void f32_to_bf16_kernel(
    const float* __restrict__ src, __bf16* __restrict__ dst, int n) {
    int i = blockIdx.x * blockDim.x + threadIdx.x;
    if (i < n) dst[i] = (__bf16)src[i];
}

// W is [K=512][N=512] row-major f32; produce Wt [N][K] bf16 so GEMM B-operand
// columns become contiguous rows.
__global__ __launch_bounds__(256) void transpose_bf16_kernel(
    const float* __restrict__ W, __bf16* __restrict__ Wt) {
    int i = blockIdx.x * blockDim.x + threadIdx.x;   // 512*512 threads
    int k = i >> 9;          // source row
    int n = i & 511;         // source col
    Wt[(size_t)n * 512 + k] = (__bf16)W[(size_t)k * 512 + n];
}

// ---------------------------------------------------------------- GEMM
// C[M,N] = (A[M,K] * Bt[N,K]^T + bias) * scale
// A, Bt bf16; output either bf16 (Cb) or f32 (Cf).
// Block 256 threads (8 waves), tile 128x128, K-step 32.
__global__ __launch_bounds__(256) void gemm_bf16_kernel(
    const __bf16* __restrict__ A, const __bf16* __restrict__ Bt,
    const float* __restrict__ bias,
    __bf16* __restrict__ Cb, float* __restrict__ Cf,
    int M, int N, int K, float scale) {
    __shared__ alignas(16) __bf16 As[128 * 40];   // padded stride 40 (80B, 16B-aligned)
    __shared__ alignas(16) __bf16 Bs[128 * 40];

    const int m0 = blockIdx.y * 128;
    const int n0 = blockIdx.x * 128;
    const int t  = threadIdx.x;
    const int w  = t >> 5;           // wave 0..7 -> rows [w*16, w*16+16)
    const int l  = t & 31;
    const int lm = l & 15;
    const int kh = l >> 4;           // lane half selects K sub-range

    v8f acc[8];
#pragma unroll
    for (int j = 0; j < 8; ++j) acc[j] = vzero8();

    const int srow = t >> 1;         // staging: 128 rows, 2 threads/row
    const int scg  = t & 1;          // 16-element column group

    for (int k0 = 0; k0 < K; k0 += 32) {
        // ---- async-stage A tile (128x32) and B tile (128 cols x 32 k)
        const __bf16* gA = A + (size_t)(m0 + srow) * K + k0 + scg * 16;
        async_load_b128(gA,     As + srow * 40 + scg * 16);
        async_load_b128(gA + 8, As + srow * 40 + scg * 16 + 8);

        const __bf16* gB = Bt + (size_t)(n0 + srow) * K + k0 + scg * 16;
        async_load_b128(gB,     Bs + srow * 40 + scg * 16);
        async_load_b128(gB + 8, Bs + srow * 40 + scg * 16 + 8);

        if (k0 + 32 < K) {
            __builtin_prefetch(gA + 32, 0, 1);
            __builtin_prefetch(gB + 32, 0, 1);
        }
        wait_async0();
        __syncthreads();

        // ---- A fragment (16x32 bf16): lane half kh holds K = kh*8..+8 and 16+kh*8..+8
        const __bf16* pa = As + (w * 16 + lm) * 40 + kh * 8;
        v16bf af = cat8(*(const v8bf*)pa, *(const v8bf*)(pa + 16));

        // ---- B fragments in groups of 4, then dense WMMA bursts
#pragma unroll
        for (int g = 0; g < 2; ++g) {
            v16bf bfrag[4];
#pragma unroll
            for (int j = 0; j < 4; ++j) {
                const __bf16* pb = Bs + ((g * 4 + j) * 16 + lm) * 40 + kh * 8;
                bfrag[j] = cat8(*(const v8bf*)pb, *(const v8bf*)(pb + 16));
            }
#pragma unroll
            for (int j = 0; j < 4; ++j) {
                acc[g * 4 + j] = __builtin_amdgcn_wmma_f32_16x16x32_bf16(
                    false, af, false, bfrag[j], (short)0, acc[g * 4 + j], false, false);
            }
        }
        __syncthreads();
    }

    // ---- epilogue: C layout VGPR r, lanes 0-15 -> M=r, lanes 16-31 -> M=r+8
#pragma unroll
    for (int j = 0; j < 8; ++j) {
        const int col = n0 + j * 16 + lm;
        const float bv = bias[col];
#pragma unroll
        for (int r = 0; r < 8; ++r) {
            const int row = m0 + w * 16 + r + 8 * kh;
            const float v = (acc[j][r] + bv) * scale;
            if (Cf) Cf[(size_t)row * N + col] = v;
            else    Cb[(size_t)row * N + col] = (__bf16)v;
        }
    }
}

// ---------------------------------------------------------------- attention
// Grid: (S/128, H, B). Block 256 = 8 waves; wave w owns 16 query rows.
// Streams 32-key tiles of K/V through LDS with online softmax.
// K tiles are DMA'd by the Tensor Data Mover (one descriptor per tile,
// issued by wave 0, completion via TENSORcnt).
__global__ __launch_bounds__(256) void flash_attn_kernel(
    const __bf16* __restrict__ Qb, const __bf16* __restrict__ Kb,
    const __bf16* __restrict__ Vb, __bf16* __restrict__ Ob) {
    constexpr int S = 4096, E = 512, Dh = 64;

    __shared__ alignas(16) __bf16 Ks[32 * Dh];      // [key][d] (TDM dense tile)
    __shared__ alignas(16) __bf16 Vts[Dh * 40];     // [d][key], padded stride
    __shared__ alignas(16) __bf16 Ps[8][16 * 32];   // per-wave P scratch

    const int    q0  = blockIdx.x * 128;
    const int    hof = blockIdx.y * Dh;
    const size_t zof = (size_t)blockIdx.z * S * E;
    const int t  = threadIdx.x;
    const int w  = t >> 5;
    const int l  = t & 31;
    const int lm = l & 15;
    const int kh = l >> 4;

    // Q fragments for this wave's 16 rows, full Dh split into two K=32 frags.
    const __bf16* qrow = Qb + zof + (size_t)(q0 + w * 16 + lm) * E + hof;
    v16bf qf[2];
#pragma unroll
    for (int f = 0; f < 2; ++f) {
        const __bf16* p = qrow + f * 32 + kh * 8;
        qf[f] = cat8(*(const v8bf*)p, *(const v8bf*)(p + 16));
    }

    v8f o[4];
#pragma unroll
    for (int d = 0; d < 4; ++d) o[d] = vzero8();
    float mrow[8], lrow[8];
#pragma unroll
    for (int r = 0; r < 8; ++r) { mrow[r] = -1e30f; lrow[r] = 0.0f; }

    const int keyt = t >> 3;   // 0..31 : staged key row
    const int seg  = t & 7;    // 8-element d group
    const unsigned ks_lds = (unsigned)(size_t)&Ks[0];

    for (int kb = 0; kb < S; kb += 32) {
        __syncthreads();   // previous iteration's LDS reads complete

        // ---- TDM-stage K tile (32 keys x 64 d, row stride E elements)
        if (t < 32) {
            const __bf16* gK0 = Kb + zof + (size_t)kb * E + hof;
            tdm_load_2d_bf16(ks_lds, gK0, Dh, 32, E);
        }

        // ---- V tile transposed into LDS via VGPRs
        const __bf16* gV = Vb + zof + (size_t)(kb + keyt) * E + hof + seg * 8;
        v8bf vv = *(const v8bf*)gV;
#pragma unroll
        for (int i = 0; i < 8; ++i) Vts[(seg * 8 + i) * 40 + keyt] = vv[i];

        if (kb + 32 < S) __builtin_prefetch(gV + 32 * E, 0, 1);

        if (t < 32) __builtin_amdgcn_s_wait_tensorcnt(0);
        __syncthreads();

        // ---- scores S = Q * K^T : 16 rows x 32 keys (two 16-key subtiles)
        v8f s0 = vzero8(), s1 = vzero8();
        {
            v16bf kf[4];   // [f*2 + j]
#pragma unroll
            for (int f = 0; f < 2; ++f) {
#pragma unroll
                for (int j = 0; j < 2; ++j) {
                    const __bf16* pb = Ks + (j * 16 + lm) * Dh + f * 32 + kh * 8;
                    kf[f * 2 + j] = cat8(*(const v8bf*)pb, *(const v8bf*)(pb + 16));
                }
            }
            s0 = __builtin_amdgcn_wmma_f32_16x16x32_bf16(
                false, qf[0], false, kf[0], (short)0, s0, false, false);
            s1 = __builtin_amdgcn_wmma_f32_16x16x32_bf16(
                false, qf[0], false, kf[1], (short)0, s1, false, false);
            s0 = __builtin_amdgcn_wmma_f32_16x16x32_bf16(
                false, qf[1], false, kf[2], (short)0, s0, false, false);
            s1 = __builtin_amdgcn_wmma_f32_16x16x32_bf16(
                false, qf[1], false, kf[3], (short)0, s1, false, false);
        }

        // ---- online softmax over 32 keys (butterfly across 16-lane halves)
#pragma unroll
        for (int r = 0; r < 8; ++r) {
            float a = s0[r], b = s1[r];
            float tm = fmaxf(a, b);
            tm = fmaxf(tm, __shfl_xor(tm, 1, 32));
            tm = fmaxf(tm, __shfl_xor(tm, 2, 32));
            tm = fmaxf(tm, __shfl_xor(tm, 4, 32));
            tm = fmaxf(tm, __shfl_xor(tm, 8, 32));
            const float mo = mrow[r];
            const float mn = fmaxf(mo, tm);
            const float alpha = __expf(mo - mn);
            const float p0 = __expf(a - mn);
            const float p1 = __expf(b - mn);
            float rs = p0 + p1;
            rs += __shfl_xor(rs, 1, 32);
            rs += __shfl_xor(rs, 2, 32);
            rs += __shfl_xor(rs, 4, 32);
            rs += __shfl_xor(rs, 8, 32);
            lrow[r] = lrow[r] * alpha + rs;
            mrow[r] = mn;
            o[0][r] *= alpha; o[1][r] *= alpha;
            o[2][r] *= alpha; o[3][r] *= alpha;
            s0[r] = p0; s1[r] = p1;
        }

        // ---- C-layout -> A-layout transpose of P through per-wave LDS
        __bf16* pw = &Ps[w][0];
#pragma unroll
        for (int r = 0; r < 8; ++r) {
            const int row = r + 8 * kh;
            pw[row * 32 + lm]      = (__bf16)s0[r];
            pw[row * 32 + 16 + lm] = (__bf16)s1[r];
        }
        __syncthreads();   // make P visible across the wave (and block-uniform)

        const __bf16* pr = pw + lm * 32 + kh * 8;
        v16bf pf = cat8(*(const v8bf*)pr, *(const v8bf*)(pr + 16));

        // ---- O += P * V : K-dim 32 keys, N-dim Dh in four 16-col subtiles
        {
            v16bf vf[4];
#pragma unroll
            for (int d = 0; d < 4; ++d) {
                const __bf16* pv = Vts + (d * 16 + lm) * 40 + kh * 8;
                vf[d] = cat8(*(const v8bf*)pv, *(const v8bf*)(pv + 16));
            }
#pragma unroll
            for (int d = 0; d < 4; ++d) {
                o[d] = __builtin_amdgcn_wmma_f32_16x16x32_bf16(
                    false, pf, false, vf[d], (short)0, o[d], false, false);
            }
        }
    }

    // ---- normalize + store concat-head layout [B,S,E] as bf16
#pragma unroll
    for (int d = 0; d < 4; ++d) {
#pragma unroll
        for (int r = 0; r < 8; ++r) {
            const float val = o[d][r] / lrow[r];
            const size_t row = (size_t)(q0 + w * 16 + r + 8 * kh);
            Ob[zof + row * E + hof + d * 16 + lm] = (__bf16)val;
        }
    }
}

// ---------------------------------------------------------------- launch
extern "C" void kernel_launch(void* const* d_in, const int* in_sizes, int n_in,
                              void* d_out, int out_size, void* d_ws, size_t ws_size,
                              hipStream_t stream) {
    (void)in_sizes; (void)n_in; (void)out_size; (void)ws_size;
    constexpr int E = 512, S = 4096, B = 2, M = B * S;

    const float* X  = (const float*)d_in[0];
    const float* Wq = (const float*)d_in[1];
    const float* bq = (const float*)d_in[2];
    const float* Wk = (const float*)d_in[3];
    const float* bk = (const float*)d_in[4];
    const float* Wv = (const float*)d_in[5];
    const float* bv = (const float*)d_in[6];
    const float* Wo = (const float*)d_in[7];
    const float* bo = (const float*)d_in[8];

    char* ws = (char*)d_ws;
    size_t off = 0;
    auto alloc = [&](size_t bytes) { char* p = ws + off; off += (bytes + 255) & ~size_t(255); return p; };
    __bf16* Xb   = (__bf16*)alloc((size_t)M * E * 2);
    __bf16* Wqt  = (__bf16*)alloc((size_t)E * E * 2);
    __bf16* Wkt  = (__bf16*)alloc((size_t)E * E * 2);
    __bf16* Wvt  = (__bf16*)alloc((size_t)E * E * 2);
    __bf16* Wot  = (__bf16*)alloc((size_t)E * E * 2);
    __bf16* Qb   = (__bf16*)alloc((size_t)M * E * 2);
    __bf16* Kbuf = (__bf16*)alloc((size_t)M * E * 2);
    __bf16* Vbuf = (__bf16*)alloc((size_t)M * E * 2);
    __bf16* Attb = (__bf16*)alloc((size_t)M * E * 2);

    // 1) convert X to bf16
    f32_to_bf16_kernel<<<(M * E + 255) / 256, 256, 0, stream>>>(X, Xb, M * E);
    // 2) transpose-convert weights
    const int wN = (E * E + 255) / 256;
    transpose_bf16_kernel<<<wN, 256, 0, stream>>>(Wq, Wqt);
    transpose_bf16_kernel<<<wN, 256, 0, stream>>>(Wk, Wkt);
    transpose_bf16_kernel<<<wN, 256, 0, stream>>>(Wv, Wvt);
    transpose_bf16_kernel<<<wN, 256, 0, stream>>>(Wo, Wot);

    // 3) Q/K/V projections (Q folded with 1/sqrt(Dh) = 1/8)
    dim3 gg(E / 128, M / 128);
    gemm_bf16_kernel<<<gg, 256, 0, stream>>>(Xb, Wqt, bq, Qb,   nullptr, M, E, E, 0.125f);
    gemm_bf16_kernel<<<gg, 256, 0, stream>>>(Xb, Wkt, bk, Kbuf, nullptr, M, E, E, 1.0f);
    gemm_bf16_kernel<<<gg, 256, 0, stream>>>(Xb, Wvt, bv, Vbuf, nullptr, M, E, E, 1.0f);

    // 4) fused flash attention
    dim3 ga(S / 128, 8, B);
    flash_attn_kernel<<<ga, 256, 0, stream>>>(Qb, Kbuf, Vbuf, Attb);

    // 5) output projection -> f32 d_out
    gemm_bf16_kernel<<<gg, 256, 0, stream>>>(Attb, Wot, bo, nullptr, (float*)d_out, M, E, E, 1.0f);
}
